// Attention_75161927680500
// MI455X (gfx1250) — matrix operand order repaired and checked by reading, not compiled, not run on hardware
//
#include <hip/hip_runtime.h>
#include <hip/hip_bf16.h>

// ---------------------------------------------------------------------------
// Multi-head attention block on gfx1250 via v_wmma_f32_16x16x32_f16.
// B=4, N=2048, C=64, HIDDEN=1024, H=16, D=64.
// ---------------------------------------------------------------------------

typedef __attribute__((ext_vector_type(16))) _Float16 v16h;
typedef __attribute__((ext_vector_type(8)))  float    v8f;

union FragH {
    v16h         h;
    unsigned int u[8];
};

__device__ __forceinline__ v8f wmma_f16(const FragH& a, const FragH& b, v8f c) {
    // D(16x16 f32) = A(16x32 f16) * B(32x16 f16) + C
    return __builtin_amdgcn_wmma_f32_16x16x32_f16(
        /*neg_a=*/false, a.h, /*neg_b=*/false, b.h,
        /*c_mod=*/(short)0, c, /*reuse_a=*/false, /*reuse_b=*/false);
}

// A-matrix (16x32 f16) per-lane K base for dword i (h=0 element), before +8*g.
// Lanes 0-15: VGPR0..3 -> K=0..7, VGPR4..7 -> K=16..23 ; lanes 16-31: +8.
__device__ __forceinline__ int a_k_base(int i) {
    return (i < 4) ? (2 * i) : (16 + 2 * (i - 4));
}

static constexpr int Bb = 4, Nn = 2048, Cc = 64, HID = 1024, Hh = 16, Dd = 64;
static constexpr int MM = Bb * Nn;            // 8192 rows

// ---------------------------------------------------------------------------
// Kernel 0: fp32 -> f16 conversions + weight transposes.
//   xh   [M][64]          = hidden_states
//   WhT  [3072][64]       = qkv_w^T
//   OWhT [64][1024]       = out_w^T
// ---------------------------------------------------------------------------
__global__ void attn_convert_kernel(const float* __restrict__ x,
                                    const float* __restrict__ qkv_w,
                                    const float* __restrict__ out_w,
                                    _Float16* __restrict__ xh,
                                    _Float16* __restrict__ WhT,
                                    _Float16* __restrict__ OWhT) {
    int id = blockIdx.x * blockDim.x + threadIdx.x;
    if (id < MM * Cc) xh[id] = (_Float16)x[id];
    if (id < 3 * HID * Cc) {                  // 196608
        int c = id >> 6, d = id & 63;
        WhT[id] = (_Float16)qkv_w[d * (3 * HID) + c];
    }
    if (id < Cc * HID) {                      // 65536
        int c = id >> 10, k = id & 1023;
        OWhT[id] = (_Float16)out_w[k * Cc + c];
    }
}

// ---------------------------------------------------------------------------
// Kernel 1: QKV projection.  [M=8192] x [K=64] x [N=3072].
// One wave computes a 16x64 strip (4 column tiles) so the A fragment is
// loaded once and reused for 8 WMMAs.  A 64-wide strip always lies inside
// one {Q|K|V} selector and one head (head dim == 64).
// Q scaled by 1/sqrt(D); Q,K stored [B,H,N,D]; V stored [B,H,D,N].
// ---------------------------------------------------------------------------
__global__ void __launch_bounds__(32)
attn_qkv_kernel(const _Float16* __restrict__ xh,
                const _Float16* __restrict__ WhT,
                const float* __restrict__ qkv_b,
                _Float16* __restrict__ Qh,
                _Float16* __restrict__ Kh,
                _Float16* __restrict__ Vt) {
    const int lane = threadIdx.x;
    const int n = lane & 15, g = lane >> 4;
    const int mtile = blockIdx.x;             // 0..511
    const int cbase = blockIdx.y * 64;        // 0..3008, step 64

    // A fragments (K = 64 -> two 16x32 fragments), reused across 4 c-tiles.
    FragH a0, a1;
    const _Float16* xrow = xh + (mtile * 16 + n) * Cc;   // A: lane = row M
#pragma unroll
    for (int i = 0; i < 8; ++i) {
        int kb = a_k_base(i) + 8 * g;
        a0.u[i] = *(const unsigned int*)(xrow + kb);
        a1.u[i] = *(const unsigned int*)(xrow + kb + 32);
    }

    const int sel = cbase >> 10;              // 0=Q 1=K 2=V (constant per wave)
    const int hh  = (cbase & 1023) >> 6;      // head (constant per wave)

#pragma unroll
    for (int dt = 0; dt < 4; ++dt) {
        const int c = cbase + dt * 16 + n;
        FragH b0, b1;
        const _Float16* wrow = WhT + c * Cc;  // B: lane = col N
#pragma unroll
        for (int j = 0; j < 8; ++j) {
            int kb = a_k_base(j) + 8 * g;
            b0.u[j] = *(const unsigned int*)(wrow + kb);
            b1.u[j] = *(const unsigned int*)(wrow + kb + 32);
        }
        v8f acc = {};
        acc = wmma_f16(a0, b0, acc);
        acc = wmma_f16(a1, b1, acc);

        const float bias = qkv_b[c];
        const int d = dt * 16 + n;            // dim within head
#pragma unroll
        for (int r = 0; r < 8; ++r) {
            int m  = mtile * 16 + r + 8 * g;  // C-layout: VGPR r -> row r+8g
            int b  = m >> 11;                 // / 2048
            int nq = m & 2047;
            float v = acc[r] + bias;
            if (sel == 0) {
                Qh[(((b * Hh + hh) * Nn + nq) * Dd) + d] = (_Float16)(v * 0.125f);
            } else if (sel == 1) {
                Kh[(((b * Hh + hh) * Nn + nq) * Dd) + d] = (_Float16)v;
            } else {
                Vt[(((b * Hh + hh) * Dd + d) * Nn) + nq] = (_Float16)v;
            }
        }
    }
}

// ---------------------------------------------------------------------------
// Kernel 2: flash-style attention.  One wave per (b,h,q-tile of 16).
// 64-key blocks: 16 WMMAs per iteration, online softmax with per-lane row
// stats (each C-tile row lives within one 16-lane half of the wave).
// ---------------------------------------------------------------------------
__global__ void __launch_bounds__(32)
attn_core_kernel(const _Float16* __restrict__ Qh,
                 const _Float16* __restrict__ Kh,
                 const _Float16* __restrict__ Vt,
                 _Float16* __restrict__ ctxh) {
    __shared__ _Float16 Plds[16 * 64];        // P tile: [q=16][k=64]

    const int lane = threadIdx.x;
    const int n = lane & 15, g = lane >> 4;
    const int bh = blockIdx.x;                // 0..63
    const int b = bh >> 4, h = bh & 15;
    const int q0 = blockIdx.y * 16;

    const _Float16* Qbase = Qh + (size_t)bh * Nn * Dd;
    const _Float16* Kbase = Kh + (size_t)bh * Nn * Dd;
    const _Float16* Vbase = Vt + (size_t)bh * Dd * Nn;

    // Q A-fragments (held in registers for the whole loop)
    FragH qa0, qa1;
    const _Float16* qrow = Qbase + (q0 + n) * Dd;
#pragma unroll
    for (int i = 0; i < 8; ++i) {
        int kb = a_k_base(i) + 8 * g;
        qa0.u[i] = *(const unsigned int*)(qrow + kb);
        qa1.u[i] = *(const unsigned int*)(qrow + kb + 32);
    }

    float mrow[8], lrow[8];
    v8f O[4];
#pragma unroll
    for (int r = 0; r < 8; ++r) { mrow[r] = -1e30f; lrow[r] = 0.0f; }
#pragma unroll
    for (int dt = 0; dt < 4; ++dt) O[dt] = (v8f){};

    for (int kb0 = 0; kb0 < Nn; kb0 += 64) {
        // Prefetch next key/value block into cache (global_prefetch_b8)
        if (kb0 + 64 < Nn) {
            __builtin_prefetch(Kbase + (kb0 + 64 + n) * Dd, 0, 3);
            __builtin_prefetch(Vbase + n * Nn + kb0 + 64, 0, 3);
        }

        // ---- scores: S[16q x 64k] as four 16x16 tiles ---------------------
        v8f S[4];
#pragma unroll
        for (int t = 0; t < 4; ++t) {
            FragH blo, bhi;
            const _Float16* krow = Kbase + (kb0 + t * 16 + n) * Dd;
#pragma unroll
            for (int j = 0; j < 8; ++j) {
                int kk = a_k_base(j) + 8 * g;
                blo.u[j] = *(const unsigned int*)(krow + kk);
                bhi.u[j] = *(const unsigned int*)(krow + kk + 32);
            }
            v8f s = {};
            s = wmma_f16(qa0, blo, s);
            s = wmma_f16(qa1, bhi, s);
            S[t] = s;
        }

        // ---- online softmax update ----------------------------------------
#pragma unroll
        for (int r = 0; r < 8; ++r) {
            float mx = fmaxf(fmaxf(S[0][r], S[1][r]), fmaxf(S[2][r], S[3][r]));
            // row reduction: row (r+8g) spans 16 lanes of one half-wave
            mx = fmaxf(mx, __shfl_xor(mx, 1, 32));
            mx = fmaxf(mx, __shfl_xor(mx, 2, 32));
            mx = fmaxf(mx, __shfl_xor(mx, 4, 32));
            mx = fmaxf(mx, __shfl_xor(mx, 8, 32));
            float mnew  = fmaxf(mrow[r], mx);
            float alpha = __expf(mrow[r] - mnew);
            float p[4], ps = 0.0f;
#pragma unroll
            for (int t = 0; t < 4; ++t) { p[t] = __expf(S[t][r] - mnew); ps += p[t]; }
            ps += __shfl_xor(ps, 1, 32);
            ps += __shfl_xor(ps, 2, 32);
            ps += __shfl_xor(ps, 4, 32);
            ps += __shfl_xor(ps, 8, 32);
            lrow[r] = lrow[r] * alpha + ps;
            mrow[r] = mnew;
#pragma unroll
            for (int dt = 0; dt < 4; ++dt) O[dt][r] *= alpha;
            int mq = r + 8 * g;               // C-layout row for this VGPR
#pragma unroll
            for (int t = 0; t < 4; ++t)
                Plds[mq * 64 + t * 16 + n] = (_Float16)p[t];
        }
        __syncthreads();                      // C-layout -> A-layout via LDS

        // ---- P A-fragments (16x64 over kseq) ------------------------------
        FragH pa0, pa1;
#pragma unroll
        for (int i = 0; i < 8; ++i) {
            int kk = a_k_base(i) + 8 * g;
            pa0.u[i] = *(const unsigned int*)(&Plds[(lane & 15) * 64 + kk]);
            pa1.u[i] = *(const unsigned int*)(&Plds[(lane & 15) * 64 + kk + 32]);
        }

        // ---- O += P * V : 4 d-tiles, K = 64 -------------------------------
#pragma unroll
        for (int dt = 0; dt < 4; ++dt) {
            FragH vlo, vhi;
            const _Float16* vrow = Vbase + (dt * 16 + n) * Nn + kb0;
#pragma unroll
            for (int j = 0; j < 8; ++j) {
                int kk = a_k_base(j) + 8 * g;
                vlo.u[j] = *(const unsigned int*)(vrow + kk);
                vhi.u[j] = *(const unsigned int*)(vrow + kk + 32);
            }
            O[dt] = wmma_f16(pa0, vlo, O[dt]);
            O[dt] = wmma_f16(pa1, vhi, O[dt]);
        }
        __syncthreads();
    }

    // ---- epilogue: normalize and store ctx [B,N,HID] as f16 ---------------
#pragma unroll
    for (int r = 0; r < 8; ++r) {
        float inv = 1.0f / lrow[r];
        int row = b * Nn + q0 + r + 8 * g;
#pragma unroll
        for (int dt = 0; dt < 4; ++dt) {
            ctxh[(size_t)row * HID + h * Dd + dt * 16 + n] =
                (_Float16)(O[dt][r] * inv);
        }
    }
}

// ---------------------------------------------------------------------------
// Kernel 3: output projection [8192 x 1024] x [1024 x 64] + bias + exact GELU.
// One wave computes the full 16x64 output strip: the A fragment of each
// k-step feeds 4 WMMAs (one per 16-wide column tile).
// ---------------------------------------------------------------------------
__global__ void __launch_bounds__(32)
attn_out_kernel(const _Float16* __restrict__ ctxh,
                const _Float16* __restrict__ OWhT,
                const float* __restrict__ out_b,
                float* __restrict__ out) {
    const int lane = threadIdx.x;
    const int n = lane & 15, g = lane >> 4;
    const int mtile = blockIdx.x;             // 0..511

    const _Float16* arow = ctxh + (size_t)(mtile * 16 + n) * HID;

    v8f acc[4];
#pragma unroll
    for (int dt = 0; dt < 4; ++dt) acc[dt] = (v8f){};

    for (int k0 = 0; k0 < HID; k0 += 32) {
        FragH a;
#pragma unroll
        for (int i = 0; i < 8; ++i) {
            int kk = a_k_base(i) + 8 * g + k0;
            a.u[i] = *(const unsigned int*)(arow + kk);
        }
#pragma unroll
        for (int dt = 0; dt < 4; ++dt) {
            FragH bf;
            const _Float16* brow = OWhT + (size_t)(dt * 16 + n) * HID + k0;
#pragma unroll
            for (int j = 0; j < 8; ++j) {
                int kk = a_k_base(j) + 8 * g;
                bf.u[j] = *(const unsigned int*)(brow + kk);
            }
            acc[dt] = wmma_f16(a, bf, acc[dt]);
        }
    }

#pragma unroll
    for (int dt = 0; dt < 4; ++dt) {
        const int c = dt * 16 + n;
        const float bias = out_b[c];
#pragma unroll
        for (int r = 0; r < 8; ++r) {
            int m = mtile * 16 + r + 8 * g;
            float x = acc[dt][r] + bias;
            float gel = 0.5f * x * (1.0f + erff(x * 0.70710678118654752f));
            out[(size_t)m * Cc + c] = gel;
        }
    }
}

// ---------------------------------------------------------------------------
// Launch
// ---------------------------------------------------------------------------
extern "C" void kernel_launch(void* const* d_in, const int* in_sizes, int n_in,
                              void* d_out, int out_size, void* d_ws, size_t ws_size,
                              hipStream_t stream) {
    const float* x      = (const float*)d_in[0];   // [4,2048,64]
    const float* qkv_w  = (const float*)d_in[1];   // [64,3072]
    const float* qkv_b  = (const float*)d_in[2];   // [3072]
    const float* out_w  = (const float*)d_in[3];   // [1024,64]
    const float* out_b  = (const float*)d_in[4];   // [64]
    float* out          = (float*)d_out;           // [4,2048,64]

    // Workspace layout (f16 staging buffers), 256B-aligned offsets
    char* ws = (char*)d_ws;
    size_t off = 0;
    auto alloc = [&](size_t bytes) {
        char* p = ws + off;
        off += (bytes + 255) & ~(size_t)255;
        return p;
    };
    _Float16* xh   = (_Float16*)alloc((size_t)MM * Cc * 2);          //  1 MB
    _Float16* WhT  = (_Float16*)alloc((size_t)3 * HID * Cc * 2);     //  384 KB
    _Float16* OWhT = (_Float16*)alloc((size_t)Cc * HID * 2);         //  128 KB
    _Float16* Qh   = (_Float16*)alloc((size_t)Bb * Hh * Nn * Dd * 2);// 16 MB
    _Float16* Kh   = (_Float16*)alloc((size_t)Bb * Hh * Nn * Dd * 2);// 16 MB
    _Float16* Vt   = (_Float16*)alloc((size_t)Bb * Hh * Dd * Nn * 2);// 16 MB
    _Float16* ctxh = (_Float16*)alloc((size_t)MM * HID * 2);         // 16 MB

    // 0) convert inputs / transpose weights to f16
    attn_convert_kernel<<<(MM * Cc + 255) / 256, 256, 0, stream>>>(
        x, qkv_w, out_w, xh, WhT, OWhT);

    // 1) QKV projection: 512 x 48 strips of 16x64, one wave each
    attn_qkv_kernel<<<dim3(MM / 16, (3 * HID) / 64), 32, 0, stream>>>(
        xh, WhT, qkv_b, Qh, Kh, Vt);

    // 2) attention core: (b*h, q-tile) = (64, 128)
    attn_core_kernel<<<dim3(Bb * Hh, Nn / 16), 32, 0, stream>>>(
        Qh, Kh, Vt, ctxh);

    // 3) output projection + bias + GELU: 512 strips of 16x64
    attn_out_kernel<<<MM / 16, 32, 0, stream>>>(
        ctxh, OWhT, out_b, out);
}
